// ch_shuffle_high_text_43198781063438
// MI455X (gfx1250) — compile-verified, matrix operand order checked by reading, not run on hardware
//
#include <hip/hip_runtime.h>
#include <math.h>

// ---------------- problem constants ----------------
#define Bz     4
#define Cc     256
#define Hh     128
#define Wp     128
#define HWp    (Hh * Wp)      // 16384
#define HEADSn 8
#define HDn    32
#define HIDn   680
#define MHn    128
#define LINn   512

typedef unsigned short u16;
typedef __attribute__((ext_vector_type(16))) __bf16 v16bf;
typedef __attribute__((ext_vector_type(8)))  float  v8f;
typedef __attribute__((ext_vector_type(4)))  float  f32x4;
typedef __attribute__((ext_vector_type(4)))  unsigned short u16x4;

// native f32 -> bf16 convert (RNE); lets the backend emit v_cvt_pk_bf16_f32
static __device__ __forceinline__ u16 f2bf(float f) {
    union { __bf16 h; u16 s; } x;
    x.h = (__bf16)f;
    return x.s;
}
static __device__ __forceinline__ float bf2f(u16 u) {
    return __uint_as_float(((unsigned int)u) << 16);
}
static __device__ __forceinline__ __bf16 bfbits(u16 u) {
    union { u16 s; __bf16 h; } x; x.s = u; return x.h;
}
static __device__ __forceinline__ u16x4 cvt4(f32x4 v) {
    u16x4 r;
    r[0] = f2bf(v[0]); r[1] = f2bf(v[1]); r[2] = f2bf(v[2]); r[3] = f2bf(v[3]);
    return r;
}

// =====================================================================
// WMMA GEMM:  Y[b] = A[b] @ X[b] (+bias[b]) (+addend)   A:(M,K) X:(K,N)
// bf16 operands, f32 accumulate.  BM=BN=128, BK=32, 256 threads/8 waves.
// Double-buffered LDS, register prefetch, branchless bounds handling.
//   MG: M may not be a multiple of 128 (A-row / store guard)
//   KG: K may not be a multiple of 32  (A-col / B-row guard)
// =====================================================================
#define LDT 40   // LDS row stride (halfs); keeps 8-half segments 16B aligned

template<bool XBF16, bool OUTBF16, bool MG, bool KG>
__global__ __launch_bounds__(256) void gemm_wmma(
    const float* __restrict__ A, size_t aStride,
    const void*  __restrict__ X,
    void*        __restrict__ Y,
    const float* __restrict__ bias, int biasStride,
    const float* __restrict__ addend,
    int M, int K, int N)
{
    __shared__ u16 As[2][128 * LDT];   // [m][k]
    __shared__ u16 Bs[2][128 * LDT];   // [n][k]  (transposed on store)

    const int tid  = threadIdx.x;
    const int lane = tid & 31;
    const int wm   = (tid >> 5) >> 1;   // 0..3
    const int wn   = (tid >> 5) & 1;    // 0..1
    const int n0   = blockIdx.x * 128;
    const int m0   = blockIdx.y * 128;
    const int bb   = blockIdx.z;

    const float* Ab = A + (size_t)bb * aStride;
    const float* Xf = (const float*)X;
    const u16*   Xh = (const u16*)X;

    // static per-thread staging coordinates (float4 granularity)
    int aRow[4], aCol[4], bRow[4], bCol[4];
#pragma unroll
    for (int i = 0; i < 4; ++i) {
        int fq = tid + i * 256;
        aRow[i] = fq >> 3;  aCol[i] = (fq & 7)  * 4;   // 128x32 A tile
        bRow[i] = fq >> 5;  bCol[i] = (fq & 31) * 4;   // 32x128 B tile
    }

    v8f acc[2][4];
#pragma unroll
    for (int i = 0; i < 2; ++i)
#pragma unroll
        for (int j = 0; j < 4; ++j)
#pragma unroll
            for (int r = 0; r < 8; ++r) acc[i][j][r] = 0.0f;

    const int lhalf = (lane >> 4) * 8;
    const f32x4 zf = {0.0f, 0.0f, 0.0f, 0.0f};
    const u16x4 zh = {0, 0, 0, 0};

    f32x4 ra[4];   // prefetched A tile
    f32x4 rbF[4];  // prefetched B tile (f32 source)
    u16x4 rbH[4];  // prefetched B tile (bf16 source)

    auto fetch = [&](int k0) {
#pragma unroll
        for (int i = 0; i < 4; ++i) {
            int gr = m0 + aRow[i];
            int gc = k0 + aCol[i];
            int rr = MG ? (gr < M ? gr : M - 1) : gr;
            int cc = KG ? (gc < K ? gc : 0)     : gc;   // K is a multiple of 4
            ra[i] = *(const f32x4*)(Ab + (size_t)rr * K + cc);
            if (MG && gr >= M) ra[i] = zf;
            if (KG && gc >= K) ra[i] = zf;
        }
#pragma unroll
        for (int i = 0; i < 4; ++i) {
            int kr = k0 + bRow[i];
            int kk = KG ? (kr < K ? kr : K - 1) : kr;
            size_t g = ((size_t)bb * K + kk) * (size_t)N + n0 + bCol[i];
            if (XBF16) {
                rbH[i] = *(const u16x4*)(Xh + g);
                if (KG && kr >= K) rbH[i] = zh;
            } else {
                rbF[i] = *(const f32x4*)(Xf + g);
                if (KG && kr >= K) rbF[i] = zf;
            }
        }
    };
    auto stage = [&](int p) {
#pragma unroll
        for (int i = 0; i < 4; ++i)
            *(u16x4*)&As[p][aRow[i] * LDT + aCol[i]] = cvt4(ra[i]);
#pragma unroll
        for (int i = 0; i < 4; ++i) {
            u16x4 h = XBF16 ? rbH[i] : cvt4(rbF[i]);
            Bs[p][(bCol[i] + 0) * LDT + bRow[i]] = h[0];
            Bs[p][(bCol[i] + 1) * LDT + bRow[i]] = h[1];
            Bs[p][(bCol[i] + 2) * LDT + bRow[i]] = h[2];
            Bs[p][(bCol[i] + 3) * LDT + bRow[i]] = h[3];
        }
    };

    const int nk = KG ? (K + 31) / 32 : (K >> 5);
    fetch(0);
    stage(0);
    __syncthreads();

    for (int it = 0; it < nk; ++it) {
        const int p = it & 1;
        if (it + 1 < nk) fetch((it + 1) * 32);   // overlap with WMMA work

        v16bf af[2], bfr[4];
#pragma unroll
        for (int i = 0; i < 2; ++i) {
            const u16* pa = &As[p][(wm * 32 + i * 16 + (lane & 15)) * LDT];
#pragma unroll
            for (int t = 0; t < 8; ++t) af[i][t]     = bfbits(pa[lhalf + t]);
#pragma unroll
            for (int t = 0; t < 8; ++t) af[i][8 + t] = bfbits(pa[16 + lhalf + t]);
        }
#pragma unroll
        for (int j = 0; j < 4; ++j) {
            const u16* pb = &Bs[p][(wn * 64 + j * 16 + (lane & 15)) * LDT];
#pragma unroll
            for (int t = 0; t < 8; ++t) bfr[j][t]     = bfbits(pb[lhalf + t]);
#pragma unroll
            for (int t = 0; t < 8; ++t) bfr[j][8 + t] = bfbits(pb[16 + lhalf + t]);
        }
#pragma unroll
        for (int i = 0; i < 2; ++i)
#pragma unroll
            for (int j = 0; j < 4; ++j)
                acc[i][j] = __builtin_amdgcn_wmma_f32_16x16x32_bf16(
                    false, af[i], false, bfr[j], (short)0, acc[i][j], false, false);

        if (it + 1 < nk) stage(1 - p);
        __syncthreads();
    }

    // ---- epilogue: bias / residual-add / f32-or-bf16 store ----
#pragma unroll
    for (int i = 0; i < 2; ++i) {
#pragma unroll
        for (int r = 0; r < 8; ++r) {
            int m = m0 + wm * 32 + i * 16 + 8 * (lane >> 4) + r;
            if (!MG || m < M) {
                float bv = bias ? bias[(size_t)bb * biasStride + m] : 0.0f;
                size_t rowg = ((size_t)bb * M + m) * (size_t)N;
#pragma unroll
                for (int j = 0; j < 4; ++j) {
                    int col = n0 + wn * 64 + j * 16 + (lane & 15);
                    float v = acc[i][j][r] + bv;
                    size_t g = rowg + col;
                    if (addend) v += addend[g];
                    if (OUTBF16) ((u16*)Y)[g] = f2bf(v);
                    else         ((float*)Y)[g] = v;
                }
            }
        }
    }
}

// =====================================================================
// text path: 2 small dense layers + full descending bitonic sort -> idx
// =====================================================================
__global__ __launch_bounds__(512) void text_k(
    const float* __restrict__ text,
    const float* __restrict__ l1w, const float* __restrict__ l1b,
    const float* __restrict__ l3w, const float* __restrict__ l3b,
    int* __restrict__ idxOut)
{
    __shared__ float t0[512];
    __shared__ float t1[512];
    __shared__ float vals[512];
    __shared__ int   ids[512];
    const int b = blockIdx.x, t = threadIdx.x;
    t0[t] = text[(size_t)b * 512 + t];
    __syncthreads();
    {
        float a = l1b[t];
        for (int i = 0; i < 512; ++i) a += t0[i] * l1w[(size_t)t * 512 + i];
        t1[t] = a;
    }
    __syncthreads();
    {
        float a = l3b[t];
        for (int i = 0; i < 512; ++i) a += t1[i] * l3w[(size_t)t * 512 + i];
        vals[t] = a; ids[t] = t;
    }
    __syncthreads();
    for (int k = 2; k <= 512; k <<= 1) {
        for (int j = k >> 1; j > 0; j >>= 1) {
            int p = t ^ j;
            if (p > t) {
                bool desc = ((t & k) == 0);
                float v0 = vals[t], v1 = vals[p];
                bool sw = desc ? (v0 < v1 || (v0 == v1 && ids[t] > ids[p]))
                               : (v0 > v1 || (v0 == v1 && ids[t] < ids[p]));
                if (sw) {
                    vals[t] = v1; vals[p] = v0;
                    int tmp = ids[t]; ids[t] = ids[p]; ids[p] = tmp;
                }
            }
            __syncthreads();
        }
    }
    idxOut[(size_t)b * 512 + t] = ids[t];
}

// =====================================================================
// W_eff[b] = conv_out_w @ conv1x1_w[idx[b],:]   (256x512x256 per batch)
// b_eff[b][o] = conv_out_w[o,:] @ conv1x1_b[idx[b]] + conv_out_b[o]
// =====================================================================
__global__ __launch_bounds__(256) void weff_k(
    const int*   __restrict__ idx,
    const float* __restrict__ cow, const float* __restrict__ cob,
    const float* __restrict__ c1w, const float* __restrict__ c1b,
    float* __restrict__ We, float* __restrict__ be)
{
    __shared__ int ids[512];
    const int o = blockIdx.x, b = blockIdx.y, t = threadIdx.x;
    ids[t]       = idx[(size_t)b * 512 + t];
    ids[256 + t] = idx[(size_t)b * 512 + 256 + t];
    __syncthreads();
    float a = 0.0f, ba = 0.0f;
    for (int j = 0; j < 512; ++j) {
        float w = cow[(size_t)o * 512 + j];
        a  += w * c1w[(size_t)ids[j] * 256 + t];
        ba += w * c1b[ids[j]];            // all lanes compute; lane 0 stores
    }
    We[((size_t)b * 256 + o) * 256 + t] = a;
    if (t == 0) be[(size_t)b * 256 + o] = ba + cob[o];
}

// =====================================================================
// LayerNorm over channel dim (C=256) per pixel.  in-place safe.
// =====================================================================
__global__ __launch_bounds__(256) void ln_chan(
    const float* __restrict__ x, float* __restrict__ y,
    const float* __restrict__ g, const float* __restrict__ be,
    int C, int N)
{
    const int p = blockIdx.x * 256 + threadIdx.x;
    const int b = blockIdx.y;
    const float* xb = x + (size_t)b * C * N;
    float s = 0.0f, s2 = 0.0f;
    for (int c = 0; c < C; ++c) {
        float v = xb[(size_t)c * N + p]; s += v; s2 += v * v;
    }
    float mu  = s / C;
    float var = s2 / C - mu * mu;
    float rs  = rsqrtf(var + 1e-5f);
    float* yb = y + (size_t)b * C * N;
    for (int c = 0; c < C; ++c) {
        float v = xb[(size_t)c * N + p];
        yb[(size_t)c * N + p] = (v - mu) * rs * g[c] + be[c];
    }
}

// =====================================================================
// per-row reduction over HW: mode 0 -> mean, mode 1 -> L2 norm
// =====================================================================
__global__ __launch_bounds__(256) void rowstat(
    const float* __restrict__ x, float* __restrict__ out, int N, int mode)
{
    __shared__ float red[256];
    const size_t row = blockIdx.x;
    const float* xr = x + row * (size_t)N;
    float a = 0.0f;
    for (int i = threadIdx.x; i < N; i += 256) {
        float v = xr[i]; a += (mode == 0) ? v : v * v;
    }
    red[threadIdx.x] = a; __syncthreads();
    for (int s = 128; s > 0; s >>= 1) {
        if (threadIdx.x < s) red[threadIdx.x] += red[threadIdx.x + s];
        __syncthreads();
    }
    if (threadIdx.x == 0) out[row] = (mode == 0) ? red[0] / N : sqrtf(red[0]);
}

// =====================================================================
// scores[b,h,c,d] = (Qraw[c]·Kraw[d]) / (||Q||·||K||) * temperature[h]
// =====================================================================
__global__ __launch_bounds__(1024) void scores_k(
    const float* __restrict__ Q, const float* __restrict__ Kr,
    const float* __restrict__ qn, const float* __restrict__ kn,
    const float* __restrict__ temp, float* __restrict__ S)
{
    __shared__ float Qs[32 * 132];
    __shared__ float Ks[32 * 132];
    const int blk = blockIdx.x, b = blk >> 3, h = blk & 7;
    const int t = threadIdx.x, c = t >> 5, d = t & 31;
    const float* Qb = Q  + ((size_t)b * 256 + h * 32) * HWp;
    const float* Kb = Kr + ((size_t)b * 256 + h * 32) * HWp;
    float acc = 0.0f;
    for (int n0 = 0; n0 < HWp; n0 += 128) {
#pragma unroll
        for (int i = 0; i < 4; ++i) {
            int idx = t + i * 1024;
            int r = idx >> 7, cc = idx & 127;
            Qs[r * 132 + cc] = Qb[(size_t)r * HWp + n0 + cc];
            Ks[r * 132 + cc] = Kb[(size_t)r * HWp + n0 + cc];
        }
        __syncthreads();
#pragma unroll 8
        for (int i = 0; i < 128; ++i) acc += Qs[c * 132 + i] * Ks[d * 132 + i];
        __syncthreads();
    }
    float qq = fmaxf(qn[(size_t)b * 256 + h * 32 + c], 1e-12f);
    float kk = fmaxf(kn[(size_t)b * 256 + h * 32 + d], 1e-12f);
    S[(((size_t)b * 8 + h) * 32 + c) * 32 + d] = acc / (qq * kk) * temp[h];
}

// =====================================================================
// mask-generator MLP + masked softmax -> probs (B,8,32,32)
// =====================================================================
__global__ __launch_bounds__(128) void mask_k(
    const float* __restrict__ opm, const float* __restrict__ spm,
    const float* __restrict__ deg,
    const float* __restrict__ fpw, const float* __restrict__ fpb,
    const float* __restrict__ ppw, const float* __restrict__ ppb,
    const float* __restrict__ m1w, const float* __restrict__ m1b,
    const float* __restrict__ m2w, const float* __restrict__ m2b,
    const float* __restrict__ S, float* __restrict__ P)
{
    __shared__ float comb[64];
    __shared__ float fin[256];
    __shared__ float h1[128];
    __shared__ float msk[1024];
    const int blk = blockIdx.x, b = blk >> 3, h = blk & 7;
    const int t = threadIdx.x;
    if (t < 32)      comb[t] = opm[(size_t)b * 256 + h * 32 + t];
    else if (t < 64) comb[t] = spm[(size_t)b * 256 + h * 32 + (t - 32)];
    __syncthreads();
    {
        float a = fpb[t];
        for (int i = 0; i < 64; ++i) a += comb[i] * fpw[t * 64 + i];
        fin[t] = a;
    }
    {
        float a = ppb[t];
        const float* dg = deg + (size_t)b * 512;
        for (int i = 0; i < 512; ++i) a += dg[i] * ppw[(size_t)t * 512 + i];
        fin[128 + t] = a;
    }
    __syncthreads();
    {
        float a = m1b[t];
        for (int i = 0; i < 256; ++i) a += fin[i] * m1w[(size_t)t * 256 + i];
        h1[t] = fmaxf(a, 0.0f);
    }
    __syncthreads();
#pragma unroll
    for (int e0 = 0; e0 < 8; ++e0) {
        int e = e0 * 128 + t;
        float a = m2b[e];
        for (int i = 0; i < 128; ++i) a += h1[i] * m2w[(size_t)e * 128 + i];
        msk[e] = 1.0f / (1.0f + expf(-a));
    }
    __syncthreads();
    if (t < 32) {
        const int c = t;
        const float* Sb = S + (((size_t)b * 8 + h) * 32 + c) * 32;
        float*       Pb = P + (((size_t)b * 8 + h) * 32 + c) * 32;
        float vals[32], mx = -3.4e38f;
        for (int d = 0; d < 32; ++d) {
            float v = Sb[d] * msk[c * 32 + d]; vals[d] = v; mx = fmaxf(mx, v);
        }
        float sm = 0.0f;
        for (int d = 0; d < 32; ++d) { float e = expf(vals[d] - mx); vals[d] = e; sm += e; }
        float inv = 1.0f / sm;
        for (int d = 0; d < 32; ++d) Pb[d] = vals[d] * inv;
    }
}

// =====================================================================
// fold probs into o_w:  Wo[b][o][h*32+d] = sum_c o_w[o][h*32+c]*P[b,h,c,d]
// =====================================================================
__global__ __launch_bounds__(256) void fuseo_k(
    const float* __restrict__ ow, const float* __restrict__ P,
    float* __restrict__ Wo)
{
    __shared__ float Ps[8192];
    const int o = blockIdx.x, b = blockIdx.y, t = threadIdx.x;
    for (int i = t; i < 8192; i += 256) Ps[i] = P[(size_t)b * 8192 + i];
    __syncthreads();
    const int h = t >> 5, d = t & 31;
    float a = 0.0f;
    for (int c = 0; c < 32; ++c)
        a += ow[(size_t)o * 256 + h * 32 + c] * Ps[(h * 32 + c) * 32 + d];
    Wo[((size_t)b * 256 + o) * 256 + t] = a;
}

// =====================================================================
// depthwise 3x3 SAME + exact GELU gate  (bf16 in, bf16 out)
// =====================================================================
__global__ __launch_bounds__(256) void dwgate_k(
    const u16* __restrict__ Ybf,
    const float* __restrict__ dww,
    u16* __restrict__ G)
{
    const int p  = blockIdx.x * 256 + threadIdx.x;
    const int ch = blockIdx.y, b = blockIdx.z;
    const int hh = p >> 7, ww = p & 127;
    const u16* y1 = Ybf + ((size_t)b * 1360 + ch) * HWp;
    const u16* y2 = Ybf + ((size_t)b * 1360 + 680 + ch) * HWp;
    const float* w1 = dww + (size_t)ch * 9;
    const float* w2 = dww + (size_t)(680 + ch) * 9;
    float a1 = 0.0f, a2 = 0.0f;
#pragma unroll
    for (int dy = -1; dy <= 1; ++dy) {
        int yy = hh + dy;
        if (yy < 0 || yy >= 128) continue;
#pragma unroll
        for (int dx = -1; dx <= 1; ++dx) {
            int xx = ww + dx;
            if (xx < 0 || xx >= 128) continue;
            int off = yy * 128 + xx;
            int wi  = (dy + 1) * 3 + (dx + 1);
            a1 += w1[wi] * bf2f(y1[off]);
            a2 += w2[wi] * bf2f(y2[off]);
        }
    }
    float g = 0.5f * a1 * (1.0f + erff(a1 * 0.70710678f)) * a2;
    G[((size_t)b * 680 + ch) * HWp + p] = f2bf(g);
}

// =====================================================================
// host-side orchestration
// =====================================================================
extern "C" void kernel_launch(void* const* d_in, const int* in_sizes, int n_in,
                              void* d_out, int out_size, void* d_ws, size_t ws_size,
                              hipStream_t stream)
{
    (void)in_sizes; (void)n_in; (void)out_size; (void)ws_size;
    const float* img  = (const float*)d_in[0];
    const float* text = (const float*)d_in[1];
    const float* l1w  = (const float*)d_in[2];
    const float* l1b  = (const float*)d_in[3];
    const float* l3w  = (const float*)d_in[4];
    const float* l3b  = (const float*)d_in[5];
    const float* c1w  = (const float*)d_in[6];
    const float* c1b  = (const float*)d_in[7];
    const float* cow  = (const float*)d_in[8];
    const float* cob  = (const float*)d_in[9];
    const float* n1w  = (const float*)d_in[10];
    const float* n1b  = (const float*)d_in[11];
    const float* n2w  = (const float*)d_in[12];
    const float* n2b  = (const float*)d_in[13];
    const float* n3w  = (const float*)d_in[14];
    const float* n3b  = (const float*)d_in[15];
    const float* qw   = (const float*)d_in[16];
    const float* kw   = (const float*)d_in[17];
    const float* vw   = (const float*)d_in[18];
    const float* ow   = (const float*)d_in[19];
    const float* temp = (const float*)d_in[20];
    const float* fpw  = (const float*)d_in[21];
    const float* fpb  = (const float*)d_in[22];
    const float* ppw  = (const float*)d_in[23];
    const float* ppb  = (const float*)d_in[24];
    const float* m1w  = (const float*)d_in[25];
    const float* m1b  = (const float*)d_in[26];
    const float* m2w  = (const float*)d_in[27];
    const float* m2b  = (const float*)d_in[28];
    const float* piw  = (const float*)d_in[29];
    const float* dww  = (const float*)d_in[30];
    const float* pow_ = (const float*)d_in[31];

    char* ws = (char*)d_ws;
    const size_t PL = (size_t)Bz * Cc * HWp * sizeof(float);   // 64 MiB
    float* QIN  = (float*)(ws + 0 * PL);
    float* KN   = (float*)(ws + 1 * PL);
    float* QRAW = (float*)(ws + 2 * PL);
    float* KRAW = (float*)(ws + 3 * PL);   // later reused for x3 = LN(att_o)
    float* VRAW = (float*)(ws + 4 * PL);
    float* ATTO = (float*)(ws + 5 * PL);
    u16* Ybf = (u16*)(ws + 6 * PL);
    const size_t ySz = (size_t)Bz * 2 * HIDn * HWp * sizeof(u16);
    u16* Gbf = (u16*)((char*)Ybf + ySz);
    const size_t gSz = (size_t)Bz * HIDn * HWp * sizeof(u16);
    char* sm = (char*)Gbf + gSz;
    int*   IDX  = (int*)sm;   sm += (size_t)Bz * 512 * sizeof(int);
    float* WEFF = (float*)sm; sm += (size_t)Bz * 256 * 256 * sizeof(float);
    float* BEFF = (float*)sm; sm += (size_t)Bz * 256 * sizeof(float);
    float* OPM  = (float*)sm; sm += (size_t)Bz * 256 * sizeof(float);
    float* SPM  = (float*)sm; sm += (size_t)Bz * 256 * sizeof(float);
    float* QN2  = (float*)sm; sm += (size_t)Bz * 256 * sizeof(float);
    float* KN2  = (float*)sm; sm += (size_t)Bz * 256 * sizeof(float);
    float* SC   = (float*)sm; sm += (size_t)Bz * 8 * 32 * 32 * sizeof(float);
    float* PR   = (float*)sm; sm += (size_t)Bz * 8 * 32 * 32 * sizeof(float);
    float* WO   = (float*)sm; sm += (size_t)Bz * 256 * 256 * sizeof(float);

    float* OUT     = (float*)d_out;
    float* OUT_IMG = OUT + (size_t)Bz * Cc * HWp;

    // 1) text routing -> full-sort indices
    text_k<<<Bz, 512, 0, stream>>>(text, l1w, l1b, l3w, l3b, IDX);
    // 2) fold shuffle + conv1x1 + conv_out into one per-batch 256x256 weight
    weff_k<<<dim3(256, Bz), 256, 0, stream>>>(IDX, cow, cob, c1w, c1b, WEFF, BEFF);
    // 3) q_in = W_eff @ img + b_eff
    gemm_wmma<false, false, false, false><<<dim3(HWp / 128, 2, Bz), 256, 0, stream>>>(
        WEFF, 256 * 256, img, QIN, BEFF, 256, nullptr, 256, 256, HWp);
    // 4) layernorms (qn in-place, kn separate)
    ln_chan<<<dim3(HWp / 256, Bz), 256, 0, stream>>>(QIN, QIN, n1w, n1b, 256, HWp);
    ln_chan<<<dim3(HWp / 256, Bz), 256, 0, stream>>>(img, KN, n2w, n2b, 256, HWp);
    // 5) channel means feeding the mask MLP
    rowstat<<<Bz * 256, 256, 0, stream>>>(KN,  OPM, HWp, 0);
    rowstat<<<Bz * 256, 256, 0, stream>>>(QIN, SPM, HWp, 0);
    // 6) Q/K/V projections
    gemm_wmma<false, false, false, false><<<dim3(HWp / 128, 2, Bz), 256, 0, stream>>>(
        qw, 0, QIN, QRAW, nullptr, 0, nullptr, 256, 256, HWp);
    gemm_wmma<false, false, false, false><<<dim3(HWp / 128, 2, Bz), 256, 0, stream>>>(
        kw, 0, KN, KRAW, nullptr, 0, nullptr, 256, 256, HWp);
    gemm_wmma<false, false, false, false><<<dim3(HWp / 128, 2, Bz), 256, 0, stream>>>(
        vw, 0, KN, VRAW, nullptr, 0, nullptr, 256, 256, HWp);
    // 7) row L2 norms (l2norm folded into score normalization)
    rowstat<<<Bz * 256, 256, 0, stream>>>(QRAW, QN2, HWp, 1);
    rowstat<<<Bz * 256, 256, 0, stream>>>(KRAW, KN2, HWp, 1);
    // 8) scores (B,8,32,32)
    scores_k<<<Bz * 8, 1024, 0, stream>>>(QRAW, KRAW, QN2, KN2, temp, SC);
    // 9) mask MLP + masked softmax -> probs
    mask_k<<<Bz * 8, 128, 0, stream>>>(OPM, SPM, text, fpw, fpb, ppw, ppb,
                                       m1w, m1b, m2w, m2b, SC, PR);
    // 10) fold probs into o_w -> per-batch 256x256 weight
    fuseo_k<<<dim3(256, Bz), 256, 0, stream>>>(ow, PR, WO);
    // 11) att_o = Wo @ Vraw  (fuses attention apply + output projection)
    gemm_wmma<false, false, false, false><<<dim3(HWp / 128, 2, Bz), 256, 0, stream>>>(
        WO, 256 * 256, VRAW, ATTO, nullptr, 0, nullptr, 256, 256, HWp);
    // 12) x3 = LN(att_o) into the now-free KRAW buffer
    ln_chan<<<dim3(HWp / 256, Bz), 256, 0, stream>>>(ATTO, KRAW, n3w, n3b, 256, HWp);
    // 13) y = pi_w @ x3  (M=1360 -> M-guarded; bf16 output to halve traffic)
    gemm_wmma<false, true, true, false><<<dim3(HWp / 128, 11, Bz), 256, 0, stream>>>(
        piw, 0, KRAW, Ybf, nullptr, 0, nullptr, 2 * HIDn, 256, HWp);
    // 14) depthwise 3x3 + exact-GELU gate (bf16 -> bf16)
    dwgate_k<<<dim3(HWp / 256, HIDn, Bz), 256, 0, stream>>>(Ybf, dww, Gbf);
    // 15) out = po_w @ gated + att_o  (bf16 operand input, K=680 -> K-guarded)
    gemm_wmma<true, false, false, true><<<dim3(HWp / 128, 2, Bz), 256, 0, stream>>>(
        pow_, 0, Gbf, OUT, nullptr, 0, ATTO, 256, HIDn, HWp);
    // 16) second tuple output: img passthrough
    hipMemcpyAsync(OUT_IMG, img, PL, hipMemcpyDeviceToDevice, stream);
}
